// ClusteringLayer_85040352460951
// MI455X (gfx1250) — compile-verified, matrix operand order checked by reading, not run on hardware
//
#include <hip/hip_runtime.h>

typedef __attribute__((ext_vector_type(16))) __bf16 v16bf;
typedef __attribute__((ext_vector_type(8)))  __bf16 bf16x8;
typedef __attribute__((ext_vector_type(8)))  float  v8f;
typedef __attribute__((ext_vector_type(4)))  unsigned int u32x4;
typedef __attribute__((ext_vector_type(8)))  int    i32x8;
typedef __attribute__((ext_vector_type(4)))  int    i32x4;

#define N_TOTAL 65536
#define K_CL    256
#define D_DIM   512
#define BM      64        // rows of z per block
#define KC      32        // K-chunk along D per WMMA pass
#define LDSROW  40        // padded LDS row stride (bf16 elems): conflict-free, 16B aligned
#define NCHUNK  (D_DIM / KC)

// ---------------------------------------------------------------------------
// Prep: clusters f32 -> bf16 hi/lo split + ||c||^2
// ---------------------------------------------------------------------------
__global__ void dec_prep_clusters(const float* __restrict__ c,
                                  float* __restrict__ c2,
                                  __bf16* __restrict__ bh,
                                  __bf16* __restrict__ bl) {
    const int k = blockIdx.x;     // cluster row
    const int t = threadIdx.x;    // 128 threads, 4 floats each
    const float4 f4 = *(const float4*)(c + (size_t)k * D_DIM + t * 4);
    float f[4] = {f4.x, f4.y, f4.z, f4.w};
    union { __bf16 b[4]; uint2 u; } uh, ul;
    float ss = 0.f;
#pragma unroll
    for (int i = 0; i < 4; ++i) {
        float v = f[i];
        ss += v * v;
        __bf16 h = (__bf16)v;
        uh.b[i] = h;
        ul.b[i] = (__bf16)(v - (float)h);
    }
    *(uint2*)(bh + (size_t)k * D_DIM + t * 4) = uh.u;
    *(uint2*)(bl + (size_t)k * D_DIM + t * 4) = ul.u;

#pragma unroll
    for (int s = 1; s < 32; s <<= 1) ss += __shfl_xor(ss, s, 32);
    __shared__ float acc;
    if (t == 0) acc = 0.f;
    __syncthreads();
    if ((t & 31) == 0) atomicAdd(&acc, ss);
    __syncthreads();
    if (t == 0) c2[k] = acc;
}

// ---------------------------------------------------------------------------
// TDM: DMA one 256x32 bf16 tile (global, row stride 512) into LDS with the
// 80B padded row layout (64B data + 16B pad -> pad_interval=3, pad_amount=3).
// D# per CDNA5 ISA ch.8 (group0 + group1, 2D tile -> groups 2/3 zero).
// amdgpu-toolchain builtin form: 6 args (g0, g1, g2, g3, extra, cpol).
// ---------------------------------------------------------------------------
__device__ __forceinline__ void tdm_load_b_tile(const __bf16* gsrc, __bf16* ldst) {
    u32x4 g0;
    g0[0] = 1u;                                     // count=1 (valid user D#)
    g0[1] = (unsigned int)(uintptr_t)ldst;          // lds_addr = low 32b of LDS ptr
    unsigned long long ga = (unsigned long long)(uintptr_t)gsrc;
    g0[2] = (unsigned int)ga;                       // global_addr[31:0]
    g0[3] = (unsigned int)(ga >> 32) | (2u << 30);  // global_addr[56:32] | type=2

    i32x8 g1;
    g1[0] = (1 << 16)      // data_size = 2 bytes
          | (1 << 20)      // pad_enable
          | (3 << 22)      // pad_interval: every 16 DWORDs (64B) of stored data
          | (3 << 25);     // pad_amount: 4 DWORDs (16B)
    g1[1] = (D_DIM & 0xFFFF) << 16;                 // tensor_dim0 = 512 (lo 16)
    g1[2] = (K_CL & 0xFFFF) << 16;                  // tensor_dim1 = 256 (lo 16)
    g1[3] = (KC & 0xFFFF) << 16;                    // tile_dim0 = 32 elems
    g1[4] = K_CL;                                   // tile_dim1 = 256 rows
    g1[5] = D_DIM;                                  // tensor_dim0_stride = 512
    g1[6] = 0;
    g1[7] = 0;

    i32x4 z4 = {0, 0, 0, 0};                        // 2D tile: groups 2/3 unused
    i32x8 z8 = {0, 0, 0, 0, 0, 0, 0, 0};
    __builtin_amdgcn_tensor_load_to_lds(g0, g1, z4, z4, z8, 0);
}

// ---------------------------------------------------------------------------
// Main: split-bf16 WMMA GEMM fused with Student-t + row normalization
// ---------------------------------------------------------------------------
__global__ __launch_bounds__(256)
void dec_q_kernel(const float* __restrict__ z,
                  const float* __restrict__ c2w,
                  const __bf16* __restrict__ bh,
                  const __bf16* __restrict__ bl,
                  float* __restrict__ out) {
    __shared__ __align__(16) __bf16 sA[2][2][BM * LDSROW];    // [stage][hi/lo]
    __shared__ __align__(16) __bf16 sB[2][2][K_CL * LDSROW];  // [stage][hi/lo]
    __shared__ float sZ2[BM];
    __shared__ float sRow[BM];
    __shared__ float sC2[K_CL];

    const int t    = threadIdx.x;
    const int lane = t & 31;
    const int wave = t >> 5;          // 0..7
    const int rt   = wave & 3;        // row tile (16 rows each)
    const int cg   = wave >> 2;       // column half: cols [cg*128, cg*128+128)
    const int nl   = lane & 15;
    const int hh   = lane >> 4;
    const size_t m0 = (size_t)blockIdx.x * BM;

    if (t < BM) { sZ2[t] = 0.f; sRow[t] = 0.f; }
    sC2[t] = c2w[t];

    // ---- A stager: z f32 -> bf16 hi/lo into LDS, + ||z||^2 partials --------
    auto loadA = [&](int stage, int ch) {
        const int r  = t >> 2;                 // 0..63: z row within block
        const int cq = t & 3;                  // 8 floats each
        const float* src = z + (m0 + r) * D_DIM + ch * KC + cq * 8;
        if (ch + 2 < NCHUNK) __builtin_prefetch(src + 2 * KC, 0, 0);
        float4 f0 = *(const float4*)src;
        float4 f1 = *(const float4*)(src + 4);
        float f[8] = {f0.x, f0.y, f0.z, f0.w, f1.x, f1.y, f1.z, f1.w};
        union { __bf16 b[8]; bf16x8 v; } uh, ul;
        float ss = 0.f;
#pragma unroll
        for (int i = 0; i < 8; ++i) {
            float v = f[i];
            ss += v * v;
            __bf16 h = (__bf16)v;
            uh.b[i] = h;
            ul.b[i] = (__bf16)(v - (float)h);
        }
        *(bf16x8*)&sA[stage][0][r * LDSROW + cq * 8] = uh.v;
        *(bf16x8*)&sA[stage][1][r * LDSROW + cq * 8] = ul.v;
        atomicAdd(&sZ2[r], ss);
    };
    // Fragment fetch matching ISA 16-bit 16x32 layout: lane m=L&15, half h=L>>4;
    // K = h*8+[0..7] then 16+h*8+[0..7]  (two 16B ds loads).
    auto frag = [&](const __bf16* base, int row) -> v16bf {
        union { v16bf v; bf16x8 h2[2]; } u;
        const __bf16* p = base + row * LDSROW + hh * 8;
        u.h2[0] = *(const bf16x8*)p;
        u.h2[1] = *(const bf16x8*)(p + 16);
        return u.v;
    };

    v8f acc[8];
#pragma unroll
    for (int j = 0; j < 8; ++j)
#pragma unroll
        for (int i = 0; i < 8; ++i) acc[j][i] = 0.f;

    __syncthreads();                  // sZ2/sRow/sC2 init visible
    if (wave == 0) {                  // TDM: B tiles for chunk 0 (hi + lo)
        tdm_load_b_tile(bh, &sB[0][0][0]);
        tdm_load_b_tile(bl, &sB[0][1][0]);
    }
    loadA(0, 0);
    if (wave == 0) __builtin_amdgcn_s_wait_tensorcnt(0);
    __syncthreads();

    for (int ch = 0; ch < NCHUNK; ++ch) {
        const int cur = ch & 1;
        if (ch + 1 < NCHUNK) {        // stage next chunk while computing this one
            if (wave == 0) {
                tdm_load_b_tile(bh + (ch + 1) * KC, &sB[cur ^ 1][0][0]);
                tdm_load_b_tile(bl + (ch + 1) * KC, &sB[cur ^ 1][1][0]);
            }
            loadA(cur ^ 1, ch + 1);
        }

        const v16bf ah = frag(sA[cur][0], rt * 16 + nl);
        const v16bf al = frag(sA[cur][1], rt * 16 + nl);
#pragma unroll
        for (int j = 0; j < 8; ++j) {
            const int n = (cg * 8 + j) * 16 + nl;
            const v16bf bhf = frag(sB[cur][0], n);
            const v16bf blf = frag(sB[cur][1], n);
            // cross ~= zh*ch + zh*cl + zl*ch   (f32 accumulate)
            acc[j] = __builtin_amdgcn_wmma_f32_16x16x32_bf16(false, ah, false, bhf,
                                                             (short)0, acc[j], false, false);
            acc[j] = __builtin_amdgcn_wmma_f32_16x16x32_bf16(false, ah, false, blf,
                                                             (short)0, acc[j], false, false);
            acc[j] = __builtin_amdgcn_wmma_f32_16x16x32_bf16(false, al, false, bhf,
                                                             (short)0, acc[j], false, false);
        }
        if (wave == 0) __builtin_amdgcn_s_wait_tensorcnt(0);
        __syncthreads();
    }

    // ---- epilogue: q = 1/(1+d2); row-normalize (ALPHA=1 -> power==1) -------
#pragma unroll
    for (int v = 0; v < 8; ++v) {
        const int m = rt * 16 + v + 8 * hh;   // C layout: lane holds N, VGPR v holds M
        const float zz = sZ2[m];
        float part = 0.f;
#pragma unroll
        for (int j = 0; j < 8; ++j) {
            const int n = (cg * 8 + j) * 16 + nl;
            float d2 = zz + sC2[n] - 2.f * acc[j][v];
            d2 = fmaxf(d2, 0.f);
            const float q = 1.f / (1.f + d2);
            acc[j][v] = q;
            part += q;
        }
#pragma unroll
        for (int s = 1; s < 16; s <<= 1) part += __shfl_xor(part, s, 32);
        if (nl == 0) atomicAdd(&sRow[m], part);
    }
    __syncthreads();
#pragma unroll
    for (int v = 0; v < 8; ++v) {
        const int m = rt * 16 + v + 8 * hh;
        const float inv = 1.f / sRow[m];
#pragma unroll
        for (int j = 0; j < 8; ++j) {
            const int n = (cg * 8 + j) * 16 + nl;
            out[(m0 + m) * K_CL + n] = acc[j][v] * inv;
        }
    }
}

// ---------------------------------------------------------------------------
extern "C" void kernel_launch(void* const* d_in, const int* in_sizes, int n_in,
                              void* d_out, int out_size, void* d_ws, size_t ws_size,
                              hipStream_t stream) {
    const float* z = (const float*)d_in[0];           // [65536, 512] f32
    const float* c = (const float*)d_in[1];           // [256, 512] f32
    float* out = (float*)d_out;                       // [65536, 256] f32

    char* ws = (char*)d_ws;                           // ~525 KB used
    float*  c2 = (float*)ws;                          // 256 f32
    __bf16* bh = (__bf16*)(ws + 1024);                // 256*512 bf16 (hi)
    __bf16* bl = (__bf16*)(ws + 1024 + (size_t)K_CL * D_DIM * 2);  // (lo)

    dec_prep_clusters<<<K_CL, 128, 0, stream>>>(c, c2, bh, bl);
    dec_q_kernel<<<N_TOTAL / BM, 256, 0, stream>>>(z, c2, bh, bl, out);
}
